// GRAPH_MAMBA_1743756722294
// MI455X (gfx1250) — matrix operand order repaired and checked by reading, not compiled, not run on hardware
//
#include <hip/hip_runtime.h>

typedef _Float16 h16;
typedef __attribute__((ext_vector_type(16))) _Float16 v16h;
typedef __attribute__((ext_vector_type(8)))  _Float16 v8h;
typedef __attribute__((ext_vector_type(8)))  float    v8f;

// Problem constants
#define BB   16
#define NN   512
#define DD   256
#define DI_  512
#define SS   16
#define RR   16
#define HH   512
#define EE   10
#define KK   3
#define PP   96
#define MROW 8192   // B*N

// ---------------------------------------------------------------------------
// Generic f16 WMMA GEMM:  out[M,N] = A[M,K] @ W^T  (W stored as [N,K] f16)
// Block = 4 waves (128 thr). Each wave: 16(M) x 64(N) strip, K step 32.
// Inner loop is branchless: all 4 N-subtiles always computed (inactive tiles
// read clamped-valid rows and their results are discarded); bounds are only
// enforced on epilogue stores. Epilogue: optional bias, silu, residual-add,
// row-scaled accumulate, f32 and/or f16 outputs, batched via blockIdx.z.
// ---------------------------------------------------------------------------
__global__ __launch_bounds__(128) void gemm16(
    const h16* __restrict__ A,  long sAb, int lda,
    const h16* __restrict__ Wt, long sWb, int ldw,
    int K, int Ncols,
    float* __restrict__ out32,  long sOb, int ldo,
    h16*  __restrict__ out16,   long sO16b, int ldo16,
    const float* __restrict__ bias,
    const float* __restrict__ resid,
    int act,
    const float* __restrict__ rowscale, int rsMod, int rsStride)
{
  const int bz   = blockIdx.z;
  const int wave = threadIdx.x >> 5;
  const int lane = threadIdx.x & 31;
  const int r    = lane & 15;
  const int half = lane >> 4;
  const int m0   = blockIdx.y * 64 + wave * 16;
  const int n0   = blockIdx.x * 64;

  v8f acc[4] = {};
  const h16* Ab = A + (size_t)bz * sAb + (size_t)(m0 + r) * lda + half * 8;
  const h16* Wb = Wt + (size_t)bz * sWb;

  const h16* Brow[4];
#pragma unroll
  for (int j = 0; j < 4; ++j) {
    int nt = n0 + j * 16;
    Brow[j] = Wb + (size_t)((nt < Ncols ? nt : 0) + r) * ldw + half * 8;
  }

#pragma unroll 2
  for (int k = 0; k < K; k += 32) {
    v8h alo = *(const v8h*)(Ab + k);
    v8h ahi = *(const v8h*)(Ab + k + 16);
    v16h af = __builtin_shufflevector(alo, ahi,
        0,1,2,3,4,5,6,7,8,9,10,11,12,13,14,15);
#pragma unroll
    for (int j = 0; j < 4; ++j) {
      v8h b0 = *(const v8h*)(Brow[j] + k);
      v8h b1 = *(const v8h*)(Brow[j] + k + 16);
      v16h bf = __builtin_shufflevector(b0, b1,
          0,1,2,3,4,5,6,7,8,9,10,11,12,13,14,15);
      acc[j] = __builtin_amdgcn_wmma_f32_16x16x32_f16(
          false, af, false, bf, (short)0, acc[j], false, false);
    }
  }

  const int c = lane & 15;
#pragma unroll
  for (int j = 0; j < 4; ++j) {
    if (n0 + j * 16 >= Ncols) continue;   // store-guard only (outside K loop)
    int col = n0 + j * 16 + c;
    float bv = bias ? bias[col] : 0.0f;
#pragma unroll
    for (int i = 0; i < 8; ++i) {
      int row = m0 + half * 8 + i;
      float v = acc[j][i] + bv;
      if (act == 1) v = v / (1.0f + __expf(-v));          // silu
      size_t oidx = (size_t)bz * sOb + (size_t)row * ldo + col;
      if (rowscale) {
        v = out32[oidx] + rowscale[(size_t)(row % rsMod) * rsStride] * v;
      } else if (resid) {
        v += resid[oidx];
      }
      if (out32) out32[oidx] = v;
      if (out16) out16[(size_t)bz * sO16b + (size_t)row * ldo16 + col] = (h16)v;
    }
  }
}

// ---------------------------------------------------------------------------
// LayerNorm over last dim (=256), emits f16 (GEMM A operand)
// ---------------------------------------------------------------------------
__global__ __launch_bounds__(256) void ln_k(const float* __restrict__ x,
    const float* __restrict__ g, const float* __restrict__ bta,
    h16* __restrict__ out16)
{
  __shared__ float s1[256], s2[256];
  int row = blockIdx.x, t = threadIdx.x;
  float v = x[(size_t)row * DD + t];
  s1[t] = v; s2[t] = v * v;
  __syncthreads();
  for (int o = 128; o > 0; o >>= 1) {
    if (t < o) { s1[t] += s1[t + o]; s2[t] += s2[t + o]; }
    __syncthreads();
  }
  float m   = s1[0] * (1.0f / DD);
  float var = s2[0] * (1.0f / DD) - m * m;
  float inv = rsqrtf(var + 1e-5f);
  out16[(size_t)row * DD + t] = (h16)((v - m) * inv * g[t] + bta[t]);
}

// ---------------------------------------------------------------------------
// Depthwise causal conv(4) + silu. Processing order p; dir=1 reads flipped.
// ---------------------------------------------------------------------------
__global__ __launch_bounds__(256) void conv_k(const float* __restrict__ inproj,
    const float* __restrict__ cw, const float* __restrict__ cb,
    float* __restrict__ xc32, h16* __restrict__ xc16, int dir)
{
  int d = blockIdx.x * 256 + threadIdx.x;   // 0..511
  int p = blockIdx.y;                       // 0..511
  int b = blockIdx.z;                       // 0..15
  float a = cb[d];
#pragma unroll
  for (int j = 0; j < 4; ++j) {
    int s = p - 3 + j;
    if (s < 0) continue;
    int n = dir ? (NN - 1 - s) : s;
    a += inproj[((size_t)b * NN + n) * 1024 + d] * cw[d * 4 + j];
  }
  a = a / (1.0f + __expf(-a));
  size_t o = ((size_t)b * NN + p) * DI_ + d;
  xc32[o] = a; xc16[o] = (h16)a;
}

// dt = softplus(dbc[:, :16] @ Wd + bd)   (K=16, VALU)
__global__ __launch_bounds__(256) void dt_k(const float* __restrict__ dbc,
    const float* __restrict__ Wd, const float* __restrict__ bd,
    float* __restrict__ dt32)
{
  int d = blockIdx.x * 256 + threadIdx.x;
  int row = blockIdx.y;
  const float* db = dbc + (size_t)row * 48;
  float a = bd[d];
#pragma unroll
  for (int rr = 0; rr < 16; ++rr) a += db[rr] * Wd[rr * DI_ + d];
  dt32[(size_t)row * DI_ + d] = (a > 20.0f) ? a : log1pf(__expf(a));
}

// Selective scan: one lane per (b,d) channel, 16 states in registers.
__global__ __launch_bounds__(256) void scan_k(const float* __restrict__ dt32,
    const float* __restrict__ xc32, const float* __restrict__ dbc,
    const float* __restrict__ Alog, float* __restrict__ y32)
{
  int d = blockIdx.x * 256 + threadIdx.x;
  int b = blockIdx.y;
  float Aexp[16];
#pragma unroll
  for (int s = 0; s < 16; ++s) Aexp[s] = __expf(Alog[(size_t)d * 16 + s]);
  float h[16];
#pragma unroll
  for (int s = 0; s < 16; ++s) h[s] = 0.0f;
  size_t rowbase = (size_t)b * NN;
  for (int p = 0; p < NN; ++p) {
    size_t row = rowbase + p;
    float dtv = dt32[row * DI_ + d];
    float xcv = xc32[row * DI_ + d];
    float dx  = dtv * xcv;
    const float* bc = dbc + row * 48 + 16;   // B at [16..31], C at [32..47]
    float yv = 0.0f;
#pragma unroll
    for (int s = 0; s < 16; ++s) {
      float dA = __expf(-Aexp[s] * dtv);
      h[s] = dA * h[s] + dx * bc[s];
      yv  += h[s] * bc[16 + s];
    }
    y32[row * DI_ + d] = yv;
  }
}

// gated[b,n,d] = (y + xc*Dp) * silu(z);  un-flips dir=1 back to original order
__global__ __launch_bounds__(256) void gate_k(const float* __restrict__ y32,
    const float* __restrict__ xc32, const float* __restrict__ inproj,
    const float* __restrict__ Dp, h16* __restrict__ gated16, int dir)
{
  int d = blockIdx.x * 256 + threadIdx.x;
  int p = blockIdx.y;
  int b = blockIdx.z;
  int n = dir ? (NN - 1 - p) : p;
  size_t rp = ((size_t)b * NN + p) * DI_ + d;
  float yv = y32[rp] + xc32[rp] * Dp[d];
  float z  = inproj[((size_t)b * NN + n) * 1024 + DI_ + d];
  float g  = z / (1.0f + __expf(-z));
  gated16[((size_t)b * NN + n) * DI_ + d] = (h16)(yv * g);
}

// sup = softmax(relu(emb @ emb^T), axis=1); emits f32 + f16 + f16-transpose
__global__ __launch_bounds__(256) void sup_k(const float* __restrict__ emb,
    float* __restrict__ sup32, h16* __restrict__ sup16,
    h16* __restrict__ supT16)
{
  __shared__ float red[256];
  int n = blockIdx.x, t = threadIdx.x;
  float en[EE];
#pragma unroll
  for (int e = 0; e < EE; ++e) en[e] = emb[n * EE + e];
  float sc[2];
#pragma unroll
  for (int q = 0; q < 2; ++q) {
    int m = t + q * 256;
    float dp = 0.0f;
#pragma unroll
    for (int e = 0; e < EE; ++e) dp += en[e] * emb[m * EE + e];
    sc[q] = dp > 0.0f ? dp : 0.0f;
  }
  red[t] = fmaxf(sc[0], sc[1]);
  __syncthreads();
  for (int o = 128; o > 0; o >>= 1) {
    if (t < o) red[t] = fmaxf(red[t], red[t + o]);
    __syncthreads();
  }
  float mx = red[0];
  __syncthreads();
  float ex0 = __expf(sc[0] - mx), ex1 = __expf(sc[1] - mx);
  red[t] = ex0 + ex1;
  __syncthreads();
  for (int o = 128; o > 0; o >>= 1) {
    if (t < o) red[t] += red[t + o];
    __syncthreads();
  }
  float inv = 1.0f / red[0];
  float v0 = ex0 * inv, v1 = ex1 * inv;
  sup32[(size_t)n * NN + t]        = v0;
  sup32[(size_t)n * NN + t + 256]  = v1;
  sup16[(size_t)n * NN + t]        = (h16)v0;
  sup16[(size_t)n * NN + t + 256]  = (h16)v1;
  supT16[(size_t)t * NN + n]       = (h16)v0;
  supT16[((size_t)t + 256) * NN + n] = (h16)v1;
}

// cheb2 = 2*(sup@sup) - I  -> f16
__global__ __launch_bounds__(256) void c2_k(const float* __restrict__ P32,
    h16* __restrict__ c216)
{
  int i = blockIdx.x, t = threadIdx.x;
#pragma unroll
  for (int q = 0; q < 2; ++q) {
    int j = t + q * 256;
    float v = 2.0f * P32[(size_t)i * NN + j] - (i == j ? 1.0f : 0.0f);
    c216[(size_t)i * NN + j] = (h16)v;
  }
}

// x[b,n,d] -> xt16[b][d,n]  (f16, W operand for cheb GEMMs)
__global__ __launch_bounds__(256) void trx_k(const float* __restrict__ x,
    h16* __restrict__ xt16)
{
  int n = blockIdx.x, b = blockIdx.y, d = threadIdx.x;
  xt16[(size_t)b * DD * NN + (size_t)d * NN + n] =
      (h16)x[((size_t)b * NN + n) * DD + d];
}

// f32 [rows,256] -> f16 with arbitrary output leading dim (for A_cat packing)
__global__ __launch_bounds__(256) void cvt16_k(const float* __restrict__ in,
    int ldi, h16* __restrict__ out, int ldo)
{
  int row = blockIdx.x, t = threadIdx.x;
  out[(size_t)row * ldo + t] = (h16)in[(size_t)row * ldi + t];
}

// xf32[b,n,o] = sum_e emb[n,e]*bias_pool[e,o]
__global__ __launch_bounds__(256) void biasinit_k(const float* __restrict__ emb,
    const float* __restrict__ bias_pool, float* __restrict__ xf32)
{
  int row = blockIdx.x, o = threadIdx.x;
  int n = row % NN;
  float a = 0.0f;
#pragma unroll
  for (int e = 0; e < EE; ++e) a += emb[n * EE + e] * bias_pool[e * DD + o];
  xf32[(size_t)row * DD + o] = a;
}

// Pack weight: src f32 [K,N] row-major -> dst f16 [N,K]
__global__ __launch_bounds__(256) void packwt_k(const float* __restrict__ src,
    h16* __restrict__ dst, int K_, int N_)
{
  int idx = blockIdx.x * 256 + threadIdx.x;
  if (idx >= K_ * N_) return;
  int n = idx / K_, k = idx - n * K_;
  dst[idx] = (h16)src[(size_t)k * N_ + n];
}

__global__ __launch_bounds__(256) void copy_k(const float* __restrict__ in,
    float* __restrict__ out, int count)
{
  int i = blockIdx.x * 256 + threadIdx.x;
  if (i < count) out[i] = in[i];
}

// ---------------------------------------------------------------------------
extern "C" void kernel_launch(void* const* d_in, const int* in_sizes, int n_in,
                              void* d_out, int out_size, void* d_ws, size_t ws_size,
                              hipStream_t stream)
{
  const float* input_   = (const float*)d_in[0];
  const float* ln1_g    = (const float*)d_in[1];
  const float* ln1_b    = (const float*)d_in[2];
  const float* ln2_g    = (const float*)d_in[3];
  const float* ln2_b    = (const float*)d_in[4];
  const float* W_in     = (const float*)d_in[5];
  const float* conv_w   = (const float*)d_in[6];
  const float* conv_b   = (const float*)d_in[7];
  const float* W_xproj  = (const float*)d_in[8];
  const float* W_dt     = (const float*)d_in[9];
  const float* b_dt     = (const float*)d_in[10];
  const float* A_log    = (const float*)d_in[11];
  const float* D_skip   = (const float*)d_in[12];
  const float* W_out    = (const float*)d_in[13];
  const float* ffn_W1   = (const float*)d_in[14];
  const float* ffn_b1   = (const float*)d_in[15];
  const float* ffn_W2   = (const float*)d_in[16];
  const float* ffn_b2   = (const float*)d_in[17];
  const float* node_emb = (const float*)d_in[18];
  const float* wpool    = (const float*)d_in[19];
  const float* bpool    = (const float*)d_in[20];
  const float* W_proj   = (const float*)d_in[21];
  const float* b_proj   = (const float*)d_in[22];
  float* outp = (float*)d_out;

  // ---- workspace carve-out ----
  char* base = (char*)d_ws;
  size_t cur = 0;
  auto alloc = [&](size_t bytes) -> char* {
    cur = (cur + 255) & ~(size_t)255;
    char* p = base + cur;
    cur += bytes;
    return p;
  };
  float* x       = (float*)alloc((size_t)MROW * DD * 4);
  h16*   xn16    = (h16*)  alloc((size_t)MROW * DD * 2);
  float* inproj0 = (float*)alloc((size_t)MROW * 1024 * 4);
  float* inproj1 = (float*)alloc((size_t)MROW * 1024 * 4);
  float* xc32    = (float*)alloc((size_t)2 * MROW * DI_ * 4);
  h16*   xc16    = (h16*)  alloc((size_t)2 * MROW * DI_ * 2);
  float* dbc     = (float*)alloc((size_t)2 * MROW * 48 * 4);
  float* dt32    = (float*)alloc((size_t)2 * MROW * DI_ * 4);
  float* y32     = (float*)alloc((size_t)2 * MROW * DI_ * 4);
  h16*   gated16 = (h16*)  alloc((size_t)MROW * DI_ * 2);
  h16*   hbuf16  = (h16*)  alloc((size_t)MROW * HH * 2);
  h16*   win_t   = (h16*)  alloc((size_t)2 * 1024 * DD * 2);
  h16*   wx_t    = (h16*)  alloc((size_t)2 * 48 * DI_ * 2);
  h16*   wo_t    = (h16*)  alloc((size_t)2 * DD * DI_ * 2);
  h16*   w1_t    = (h16*)  alloc((size_t)HH * DD * 2);
  h16*   w2_t    = (h16*)  alloc((size_t)DD * HH * 2);
  h16*   wproj_t = (h16*)  alloc((size_t)PP * DD * 2);
  h16*   wt_e    = (h16*)  alloc((size_t)EE * DD * (KK * DD) * 2);
  float* sup32   = (float*)alloc((size_t)NN * NN * 4);
  h16*   sup16   = (h16*)  alloc((size_t)NN * NN * 2);
  h16*   supT16  = (h16*)  alloc((size_t)NN * NN * 2);
  float* P32     = (float*)alloc((size_t)NN * NN * 4);
  h16*   c216    = (h16*)  alloc((size_t)NN * NN * 2);
  h16*   xt16    = (h16*)  alloc((size_t)BB * DD * NN * 2);
  h16*   Acat    = (h16*)  alloc((size_t)MROW * (KK * DD) * 2);
  float* xf32    = (float*)alloc((size_t)MROW * DD * 4);
  h16*   xf16    = (h16*)  alloc((size_t)MROW * DD * 2);
  if (cur > ws_size) return;  // workspace too small; bail deterministically

  // ---- pack weights to transposed f16 ----
  for (int dir = 0; dir < 2; ++dir) {
    packwt_k<<<(256 * 1024 + 255) / 256, 256, 0, stream>>>(
        W_in + (size_t)dir * 256 * 1024, win_t + (size_t)dir * 1024 * 256, 256, 1024);
    packwt_k<<<(512 * 48 + 255) / 256, 256, 0, stream>>>(
        W_xproj + (size_t)dir * 512 * 48, wx_t + (size_t)dir * 48 * 512, 512, 48);
    packwt_k<<<(512 * 256 + 255) / 256, 256, 0, stream>>>(
        W_out + (size_t)dir * 512 * 256, wo_t + (size_t)dir * 256 * 512, 512, 256);
  }
  packwt_k<<<(256 * 512 + 255) / 256, 256, 0, stream>>>(ffn_W1, w1_t, 256, 512);
  packwt_k<<<(512 * 256 + 255) / 256, 256, 0, stream>>>(ffn_W2, w2_t, 512, 256);
  packwt_k<<<(256 * 96 + 255) / 256, 256, 0, stream>>>(W_proj, wproj_t, 256, 96);
  for (int e = 0; e < EE; ++e)
    packwt_k<<<(768 * 256 + 255) / 256, 256, 0, stream>>>(
        wpool + (size_t)e * 768 * 256, wt_e + (size_t)e * 256 * 768, 768, 256);

  copy_k<<<(MROW * DD + 255) / 256, 256, 0, stream>>>(input_, x, MROW * DD);

  // ---- layers ----
  for (int layer = 0; layer < 3; ++layer) {
    ln_k<<<MROW, 256, 0, stream>>>(x, ln1_g, ln1_b, xn16);
    for (int dir = 0; dir < 2; ++dir) {
      float* inproj = dir ? inproj1 : inproj0;
      float* xcf = xc32 + (size_t)dir * MROW * DI_;
      h16*   xch = xc16 + (size_t)dir * MROW * DI_;
      float* dbcd = dbc + (size_t)dir * MROW * 48;
      float* dtd  = dt32 + (size_t)dir * MROW * DI_;
      float* yd   = y32 + (size_t)dir * MROW * DI_;
      // in-proj: [8192,256]@[256,1024]
      gemm16<<<dim3(16, 128, 1), 128, 0, stream>>>(
          xn16, 0, 256, win_t + (size_t)dir * 1024 * 256, 0, 256, 256, 1024,
          inproj, 0, 1024, nullptr, 0, 1, nullptr, nullptr, 0, nullptr, 1, 0);
      conv_k<<<dim3(2, 512, 16), 256, 0, stream>>>(
          inproj, conv_w + (size_t)dir * DI_ * 4, conv_b + (size_t)dir * DI_,
          xcf, xch, dir);
      // x-proj: [8192,512]@[512,48]
      gemm16<<<dim3(1, 128, 1), 128, 0, stream>>>(
          xch, 0, 512, wx_t + (size_t)dir * 48 * 512, 0, 512, 512, 48,
          dbcd, 0, 48, nullptr, 0, 1, nullptr, nullptr, 0, nullptr, 1, 0);
      dt_k<<<dim3(2, MROW), 256, 0, stream>>>(
          dbcd, W_dt + (size_t)dir * 16 * DI_, b_dt + (size_t)dir * DI_, dtd);
      scan_k<<<dim3(2, 16), 256, 0, stream>>>(
          dtd, xcf, dbcd, A_log + (size_t)dir * DI_ * SS, yd);
      gate_k<<<dim3(2, 512, 16), 256, 0, stream>>>(
          yd, xcf, inproj, D_skip + (size_t)dir * DI_, gated16, dir);
      // out-proj + residual into x: [8192,512]@[512,256]
      gemm16<<<dim3(4, 128, 1), 128, 0, stream>>>(
          gated16, 0, 512, wo_t + (size_t)dir * 256 * 512, 0, 512, 512, 256,
          x, 0, 256, nullptr, 0, 1, nullptr, x, 0, nullptr, 1, 0);
    }
    // FFN
    ln_k<<<MROW, 256, 0, stream>>>(x, ln2_g, ln2_b, xn16);
    gemm16<<<dim3(8, 128, 1), 128, 0, stream>>>(           // silu(xn@W1+b1)->f16
        xn16, 0, 256, w1_t, 0, 256, 256, 512,
        nullptr, 0, 1, hbuf16, 0, 512, ffn_b1, nullptr, 1, nullptr, 1, 0);
    gemm16<<<dim3(4, 128, 1), 128, 0, stream>>>(           // x += h@W2+b2
        hbuf16, 0, 512, w2_t, 0, 512, 512, 256,
        x, 0, 256, nullptr, 0, 1, ffn_b2, x, 0, nullptr, 1, 0);
  }

  // ---- graph section ----
  sup_k<<<NN, 256, 0, stream>>>(node_emb, sup32, sup16, supT16);
  gemm16<<<dim3(8, 8, 1), 128, 0, stream>>>(               // P = sup@sup
      sup16, 0, 512, supT16, 0, 512, 512, 512,
      P32, 0, 512, nullptr, 0, 1, nullptr, nullptr, 0, nullptr, 1, 0);
  c2_k<<<NN, 256, 0, stream>>>(P32, c216);
  trx_k<<<dim3(512, 16), 256, 0, stream>>>(x, xt16);
  cvt16_k<<<MROW, 256, 0, stream>>>(x, 256, Acat, 768);    // xg0 slot
  // xg1 = sup @ x[b]   -> Acat[:, 256:512]
  gemm16<<<dim3(4, 8, 16), 128, 0, stream>>>(
      sup16, 0, 512, xt16, (long)DD * NN, 512, 512, 256,
      nullptr, 0, 1, Acat + 256, (long)NN * 768, 768,
      nullptr, nullptr, 0, nullptr, 1, 0);
  // xg2 = cheb2 @ x[b] -> Acat[:, 512:768]
  gemm16<<<dim3(4, 8, 16), 128, 0, stream>>>(
      c216, 0, 512, xt16, (long)DD * NN, 512, 512, 256,
      nullptr, 0, 1, Acat + 512, (long)NN * 768, 768,
      nullptr, nullptr, 0, nullptr, 1, 0);
  biasinit_k<<<MROW, 256, 0, stream>>>(node_emb, bpool, xf32);
  for (int e = 0; e < EE; ++e) {
    // xf32 += emb[n,e] * (Acat @ Wpool_e)
    gemm16<<<dim3(4, 128, 1), 128, 0, stream>>>(
        Acat, 0, 768, wt_e + (size_t)e * 256 * 768, 0, 768, 768, 256,
        xf32, 0, 256, nullptr, 0, 1, nullptr, nullptr, 0,
        node_emb + e, NN, EE);
  }
  cvt16_k<<<MROW, 256, 0, stream>>>(xf32, 256, xf16, 256);
  // out = xf @ W_proj + b_proj   [8192,256]@[256,96]
  gemm16<<<dim3(2, 128, 1), 128, 0, stream>>>(
      xf16, 0, 256, wproj_t, 0, 256, 256, 96,
      outp, 0, 96, nullptr, 0, 1, b_proj, nullptr, 0, nullptr, 1, 0);
}
